// LeNetOri_24515673325688
// MI455X (gfx1250) — matrix operand order repaired and checked by reading, not compiled, not run on hardware
//
#include <hip/hip_runtime.h>
#include <hip/hip_bf16.h>

// MI455X / gfx1250, wave32. All matmuls via v_wmma_f32_16x16x32_bf16.
// Branch-free A-fragment gathers: B operands are zero-padded in K, so A
// elements at padded k are multiplied by 0 -> clamp index, load garbage.
//
// ws layout (bytes):
//   Wb1  @ 0        : 16x32   bf16
//   Wc3  @ 1024     : 16x160  bf16
//   Wc5  @ 6144     : 128x416 bf16
//   Wf6  @ 112640   : 96x128  bf16
//   Wout @ 137216   : 16x96   bf16
//   P1   @ 140288   : B*1176  bf16
//   P2   @ +B*2352  : B*416   bf16
// total ~52.4 MB.

typedef __attribute__((ext_vector_type(16))) __bf16 v16bf;
typedef __attribute__((ext_vector_type(8)))  __bf16 v8bf;
typedef __attribute__((ext_vector_type(8)))  float  v8f;

#define WMMA_BF16(a,b,c) \
  __builtin_amdgcn_wmma_f32_16x16x32_bf16(false,(a),false,(b),(short)0,(c),false,false)

// tanh(x) = sign(x) * (1 - 2/(e^{2|x|}+1)).
// One v_exp_f32 + one raw v_rcp_f32 (NOT the IEEE div expansion) + fma.
__device__ __forceinline__ float fast_tanhf(float x){
  float e2 = __expf(2.0f * fabsf(x));
  float t  = __builtin_fmaf(-2.0f, __builtin_amdgcn_rcpf(e2 + 1.0f), 1.0f);
  return copysignf(t, x);
}

__device__ __forceinline__ v8f vzero8(){
  v8f z = {0.f,0.f,0.f,0.f,0.f,0.f,0.f,0.f};
  return z;
}

// B fragment (32x16): lane = column N (L%16), 16 contiguous k at
// k = kb + 16*(L/16) + e.  Two b128 loads.
__device__ __forceinline__ v16bf load_bfrag(const __bf16* p){
  v8bf lo = *reinterpret_cast<const v8bf*>(p);
  v8bf hi = *reinterpret_cast<const v8bf*>(p + 8);
  v16bf r;
#pragma unroll
  for (int i = 0; i < 8; ++i){ r[i] = lo[i]; r[i+8] = hi[i]; }
  return r;
}

// A fragment (16x32) from a row-major row: lane row m = L%16, g = L/16,
// 16 contiguous k at k = kb + 16g + e.
__device__ __forceinline__ v16bf load_afrag(const __bf16* row, int kb, int g){
  v8bf lo = *reinterpret_cast<const v8bf*>(row + kb + g*16);
  v8bf hi = *reinterpret_cast<const v8bf*>(row + kb + g*16 + 8);
  v16bf r;
#pragma unroll
  for (int i = 0; i < 8; ++i){ r[i] = lo[i]; r[i+8] = hi[i]; }
  return r;
}

// A-layout k index for element e in lane-half g (compile-time after unroll).
__device__ __forceinline__ int afrag_k(int e, int g){
  return g*16 + e;
}

// ---------------------------------------------------------------------------
// Weight repack: zero-padded bf16 GEMM-B operands [N][Kpad].
// ---------------------------------------------------------------------------
__global__ void __launch_bounds__(256)
k_prep(const float* __restrict__ c1_w, const float* __restrict__ fs_w,
       const float* __restrict__ sc_w, const float* __restrict__ th_w,
       const float* __restrict__ fo_w, const float* __restrict__ c5_w,
       const float* __restrict__ f6_w, const float* __restrict__ out_w,
       __bf16* __restrict__ Wb1, __bf16* __restrict__ Wc3,
       __bf16* __restrict__ Wc5, __bf16* __restrict__ Wf6,
       __bf16* __restrict__ Wout)
{
  const int i = blockIdx.x * 256 + threadIdx.x;
  if (i < 512){                                    // Wb1: 16x32
    const int n = i / 32, k = i % 32;
    float v = (n < 6 && k < 25) ? c1_w[n*25 + k] : 0.f;
    Wb1[i] = (__bf16)v;
  }
  const int j = i - 512;                           // Wc3: 16x160 (fold grouping)
  if (j >= 0 && j < 2560){
    const int oc = j / 160, k = j % 160;
    float v = 0.f;
    if (k < 150){
      const int ci = k / 25, rc = k % 25;
      int ch[6]; int len; const float* w;
      if (oc < 6){        len = 3; w = fs_w; for (int t=0;t<3;t++) ch[t] = (oc + t) % 6; }
      else if (oc < 12){  len = 4; w = sc_w; const int ii = oc - 6;
                          for (int t=0;t<4;t++) ch[t] = (ii + t) % 6; }
      else if (oc < 15){  len = 4; w = th_w; const int ii = oc - 12;
                          const int offs[4] = {0,1,3,4};
                          for (int t=0;t<4;t++) ch[t] = (ii + offs[t]) % 6; }
      else {              len = 6; w = fo_w; for (int t=0;t<6;t++) ch[t] = t; }
      for (int t = 0; t < len; ++t) if (ch[t] == ci) v = w[t*25 + rc];
    }
    Wc3[j] = (__bf16)v;
  }
  const int m = i - (512 + 2560);                  // Wc5: 128x416
  if (m >= 0 && m < 128*416){
    const int n = m / 416, k = m % 416;
    float v = (n < 120 && k < 400) ? c5_w[n*400 + k] : 0.f;
    Wc5[m] = (__bf16)v;
  }
  const int p = i - (512 + 2560 + 53248);          // Wf6: 96x128
  if (p >= 0 && p < 96*128){
    const int n = p / 128, k = p % 128;
    float v = (n < 84 && k < 120) ? f6_w[n*120 + k] : 0.f;
    Wf6[p] = (__bf16)v;
  }
  const int q = i - (512 + 2560 + 53248 + 12288);  // Wout: 16x96
  if (q >= 0 && q < 16*96){
    const int n = q / 96, k = q % 96;
    float v = (n < 10 && k < 84) ? out_w[n*84 + k] : 0.f;
    Wout[q] = (__bf16)v;
  }
}

// ---------------------------------------------------------------------------
// conv1(5x5) + tanh + avgpool2, implicit GEMM. Block = 1 image, 4 waves.
// M = 16 output columns of one row, N = 16 (6 real ch), K = 32 (25 real).
// Gather offsets hoisted; all LDS ops branch-free.
// ---------------------------------------------------------------------------
__global__ void __launch_bounds__(128)
k_conv1(const float* __restrict__ x, const __bf16* __restrict__ Wb1,
        const float* __restrict__ c1_b, __bf16* __restrict__ P1)
{
  __shared__ alignas(16) __bf16 xs[32][32];
  __shared__ alignas(16) __bf16 conv[16][28][32];   // padded: stores need no guards
  const int b = blockIdx.x, tid = threadIdx.x;
  for (int i = tid; i < 1024; i += 128)
    (&xs[0][0])[i] = (__bf16)x[(size_t)b*1024 + i];
  __syncthreads();

  const int lane = tid & 31, wave = tid >> 5;
  const int g = lane >> 4, ml = lane & 15;
  const v16bf bw = load_bfrag(Wb1 + ml*32 + g*16);
  const float bias = c1_b[ml < 6 ? ml : 5];         // pad rows never pooled

  // per-lane gather offsets (halfword units), loop-invariant across tiles
  int off[16];
#pragma unroll
  for (int e = 0; e < 16; ++e){
    int k = afrag_k(e, g);
    if (k > 24) k = 24;                             // Wb1 rows k>=25 are zero
    off[e] = (k / 5) * 32 + (k % 5);
  }

  for (int t = wave; t < 56; t += 4){               // 28 rows x 2 col-tiles
    const int oh  = t >> 1, owb = (t & 1) * 16;
    const int ow  = (owb + ml < 28) ? owb + ml : 27;
    const __bf16* base = &xs[oh][ow];
    v16bf a;
#pragma unroll
    for (int e = 0; e < 16; ++e) a[e] = base[off[e]];
    v8f acc = vzero8();
    acc = WMMA_BF16(a, bw, acc);
#pragma unroll
    for (int r = 0; r < 8; ++r)                     // D: M = r+8g, N = ml
      conv[ml][oh][owb + r + 8*g] = (__bf16)fast_tanhf(acc[r] + bias);
  }
  __syncthreads();

  for (int o = tid; o < 6*14*14; o += 128){         // fused 2x2 avg pool
    const int ch = o / 196, rem = o % 196, ph = rem / 14, pw = rem % 14;
    float s = (float)conv[ch][2*ph][2*pw]   + (float)conv[ch][2*ph][2*pw+1]
            + (float)conv[ch][2*ph+1][2*pw] + (float)conv[ch][2*ph+1][2*pw+1];
    P1[(size_t)b*1176 + o] = (__bf16)(s * 0.25f);
  }
}

// ---------------------------------------------------------------------------
// C3 grouped convs + tanh + avgpool2 as ONE dense implicit GEMM.
// M = 16 of 100 positions, N = 16 out channels, K = 160 (zero-masked groups).
// K-step outer, both of a wave's position tiles inner (offset math amortized).
// ---------------------------------------------------------------------------
__global__ void __launch_bounds__(128)
k_c3(const __bf16* __restrict__ P1, const __bf16* __restrict__ Wc3,
     const float* __restrict__ fs_b, const float* __restrict__ sc_b,
     const float* __restrict__ th_b, const float* __restrict__ fo_b,
     __bf16* __restrict__ P2)
{
  __shared__ alignas(16) __bf16 img[6][14][14];
  __shared__ alignas(16) __bf16 conv3[16][112];     // padded: no store guards
  __shared__ float biases[16];
  const int b = blockIdx.x, tid = threadIdx.x;
  for (int i = tid; i < 1176; i += 128)
    (&img[0][0][0])[i] = P1[(size_t)b*1176 + i];
  if (tid < 16)
    biases[tid] = (tid < 6) ? fs_b[0] : (tid < 12) ? sc_b[0]
                : (tid < 15) ? th_b[0] : fo_b[0];
  __syncthreads();

  const int lane = tid & 31, wave = tid >> 5;
  const int g = lane >> 4, ml = lane & 15;
  v16bf bw[5];
#pragma unroll
  for (int s = 0; s < 5; ++s)
    bw[s] = load_bfrag(Wc3 + ml*160 + s*32 + g*16);
  const float bias = biases[ml];

  // this wave's two position tiles (wave 3's 2nd clamps to tile 6: benign
  // duplicate write of identical values)
  const int pta = wave, ptb = (wave + 4 < 7) ? wave + 4 : 6;
  const int pA = (pta*16 + ml < 100) ? pta*16 + ml : 99;
  const int pB = (ptb*16 + ml < 100) ? ptb*16 + ml : 99;
  const __bf16* baseA = &img[0][pA / 10][pA % 10];
  const __bf16* baseB = &img[0][pB / 10][pB % 10];

  v8f accA = vzero8(), accB = vzero8();
#pragma unroll
  for (int s = 0; s < 5; ++s){
    int off[16];
#pragma unroll
    for (int e = 0; e < 16; ++e){
      int k = s*32 + afrag_k(e, g);
      if (k > 149) k = 149;                         // Wc3 rows k>=150 are zero
      const int ci = k / 25, rc = k % 25;
      off[e] = ci*196 + (rc / 5)*14 + (rc % 5);
    }
    v16bf a, a2;
#pragma unroll
    for (int e = 0; e < 16; ++e){ a[e] = baseA[off[e]]; a2[e] = baseB[off[e]]; }
    accA = WMMA_BF16(a,  bw[s], accA);
    accB = WMMA_BF16(a2, bw[s], accB);
  }
#pragma unroll
  for (int r = 0; r < 8; ++r){
    conv3[ml][pta*16 + r + 8*g] = (__bf16)fast_tanhf(accA[r] + bias);
    conv3[ml][ptb*16 + r + 8*g] = (__bf16)fast_tanhf(accB[r] + bias);
  }
  __syncthreads();

  for (int o = tid; o < 400; o += 128){             // pool -> conv5 K layout
    const int oc = o / 25, pos = o % 25, ph = pos / 5, pw = pos % 5;
    float s = (float)conv3[oc][(2*ph)*10 + 2*pw]   + (float)conv3[oc][(2*ph)*10 + 2*pw + 1]
            + (float)conv3[oc][(2*ph+1)*10 + 2*pw] + (float)conv3[oc][(2*ph+1)*10 + 2*pw + 1];
    P2[(size_t)b*416 + o] = (__bf16)(s * 0.25f);
  }
  if (tid < 16) P2[(size_t)b*416 + 400 + tid] = (__bf16)0.f;  // K padding
}

// ---------------------------------------------------------------------------
// conv5(416->128) -> tanh -> f6(128->96) -> tanh -> out(96->16), fused WMMA.
// 1 wave per 16-row batch tile; f32 D re-staged through LDS as bf16 A.
// Pad columns are annihilated by next layer's zero K-padding, so bias
// indices just clamp (no lane guards).
// ---------------------------------------------------------------------------
__global__ void __launch_bounds__(128)
k_mlp(const __bf16* __restrict__ P2, const __bf16* __restrict__ Wc5,
      const __bf16* __restrict__ Wf6, const __bf16* __restrict__ Wout,
      const float* __restrict__ c5_b, const float* __restrict__ f6_b,
      const float* __restrict__ out_b, float* __restrict__ out)
{
  __shared__ alignas(16) __bf16 T1[4][16][128];
  __shared__ alignas(16) __bf16 T2[4][16][96];
  const int tid = threadIdx.x, wave = tid >> 5, lane = tid & 31;
  const int g = lane >> 4, ml = lane & 15;
  const int bb = (blockIdx.x * 4 + wave) * 16;
  const __bf16* arow = P2 + (size_t)(bb + ml) * 416;
  __builtin_prefetch(Wc5 + ml*416, 0, 0);           // global_prefetch_b8

  // --- conv5: K=416 (13 steps), 8 N-tiles ---
  v8f acc[8];
#pragma unroll
  for (int t = 0; t < 8; ++t) acc[t] = vzero8();
  for (int kb = 0; kb < 416; kb += 32){
    const v16bf a = load_afrag(arow, kb, g);
#pragma unroll
    for (int t = 0; t < 8; ++t){
      const v16bf bwv = load_bfrag(Wc5 + (size_t)(t*16 + ml)*416 + kb + g*16);
      acc[t] = WMMA_BF16(a, bwv, acc[t]);
    }
  }
#pragma unroll
  for (int t = 0; t < 8; ++t){
    const int n = t*16 + ml;
    const float bias = c5_b[n < 120 ? n : 119];
#pragma unroll
    for (int r = 0; r < 8; ++r)
      T1[wave][r + 8*g][n] = (__bf16)fast_tanhf(acc[t][r] + bias);
  }
  __syncthreads();

  // --- f6: K=128 (4 steps), 6 N-tiles ---
  v8f acc2[6];
#pragma unroll
  for (int t = 0; t < 6; ++t) acc2[t] = vzero8();
  for (int kb = 0; kb < 128; kb += 32){
    const v16bf a = load_afrag(&T1[wave][ml][0], kb, g);
#pragma unroll
    for (int t = 0; t < 6; ++t){
      const v16bf bwv = load_bfrag(Wf6 + (size_t)(t*16 + ml)*128 + kb + g*16);
      acc2[t] = WMMA_BF16(a, bwv, acc2[t]);
    }
  }
#pragma unroll
  for (int t = 0; t < 6; ++t){
    const int n = t*16 + ml;
    const float bias = f6_b[n < 84 ? n : 83];
#pragma unroll
    for (int r = 0; r < 8; ++r)
      T2[wave][r + 8*g][n] = (__bf16)fast_tanhf(acc2[t][r] + bias);
  }
  __syncthreads();

  // --- out: K=96 (3 steps), 1 N-tile ---
  v8f acc3 = vzero8();
  for (int kb = 0; kb < 96; kb += 32){
    const v16bf a   = load_afrag(&T2[wave][ml][0], kb, g);
    const v16bf bwv = load_bfrag(Wout + (size_t)ml*96 + kb + g*16);
    acc3 = WMMA_BF16(a, bwv, acc3);
  }
  if (ml < 10){
    const float ob = out_b[ml];
#pragma unroll
    for (int r = 0; r < 8; ++r)
      out[(size_t)(bb + r + 8*g)*10 + ml] = acc3[r] + ob;
  }
}

// ---------------------------------------------------------------------------
extern "C" void kernel_launch(void* const* d_in, const int* in_sizes, int n_in,
                              void* d_out, int out_size, void* d_ws, size_t ws_size,
                              hipStream_t stream)
{
  (void)n_in; (void)out_size; (void)ws_size;
  const float* x     = (const float*)d_in[0];
  const float* c1_w  = (const float*)d_in[1];
  const float* c1_b  = (const float*)d_in[2];
  const float* fs_w  = (const float*)d_in[3];
  const float* fs_b  = (const float*)d_in[4];
  const float* sc_w  = (const float*)d_in[5];
  const float* sc_b  = (const float*)d_in[6];
  const float* th_w  = (const float*)d_in[7];
  const float* th_b  = (const float*)d_in[8];
  const float* fo_w  = (const float*)d_in[9];
  const float* fo_b  = (const float*)d_in[10];
  const float* c5_w  = (const float*)d_in[11];
  const float* c5_b  = (const float*)d_in[12];
  const float* f6_w  = (const float*)d_in[13];
  const float* f6_b  = (const float*)d_in[14];
  const float* out_w = (const float*)d_in[15];
  const float* out_b = (const float*)d_in[16];
  float* out = (float*)d_out;
  const int B = in_sizes[0] / 1024;                 // 16384

  char* ws = (char*)d_ws;
  __bf16* Wb1  = (__bf16*)(ws + 0);
  __bf16* Wc3  = (__bf16*)(ws + 1024);
  __bf16* Wc5  = (__bf16*)(ws + 6144);
  __bf16* Wf6  = (__bf16*)(ws + 112640);
  __bf16* Wout = (__bf16*)(ws + 137216);
  __bf16* P1   = (__bf16*)(ws + 140288);
  __bf16* P2   = (__bf16*)(ws + 140288 + (size_t)B * 1176 * 2);

  k_prep<<<(70144 + 255)/256, 256, 0, stream>>>(c1_w, fs_w, sc_w, th_w, fo_w,
                                                c5_w, f6_w, out_w,
                                                Wb1, Wc3, Wc5, Wf6, Wout);
  k_conv1<<<B, 128, 0, stream>>>(x, Wb1, c1_b, P1);
  k_c3<<<B, 128, 0, stream>>>(P1, Wc3, fs_b, sc_b, th_b, fo_b, P2);
  k_mlp<<<B / 64, 128, 0, stream>>>(P2, Wc5, Wf6, Wout, c5_b, f6_b, out_b, out);
}
